// BatchGraphAttention_84378927497895
// MI455X (gfx1250) — matrix operand order repaired and checked by reading, not compiled, not run on hardware
//
#include <hip/hip_runtime.h>
#include <hip/hip_bf16.h>

// GAT layer fused kernels for gfx1250 (MI455X), wave32 + WMMA.
// B=4, N=2048, F=64, H=4, F_=64.

typedef __attribute__((ext_vector_type(16))) _Float16 v16h;
typedef __attribute__((ext_vector_type(8)))  float    v8f;
typedef __attribute__((ext_vector_type(4)))  float    v4f;

#define GB  4
#define GN  2048
#define GF  64
#define GH  4
#define GFO 64
#define NT16 (GN / 16)   // 128 row tiles

__device__ __forceinline__ float leaky02(float x) {
    return x >= 0.0f ? x : 0.2f * x;
}

// ---------------------------------------------------------------------------
// Kernel 1: lin = X @ W per (b,h) via WMMA f32_16x16x32_f16.
// Outputs:
//   linT  : [B,H,F_,N] f16  (transposed so phase-2 B-operand loads are contiguous)
//   s_self, s_neigh : [B,H,N] f32 (row dot products with a_self / a_neigh)
// Grid: B*H*8 blocks x 128 threads (4 waves). Wave-slot g handles mtiles g, g+32, ...
// ---------------------------------------------------------------------------
__global__ __launch_bounds__(128) void gat_lin_kernel(
    const float* __restrict__ X, const float* __restrict__ W,
    const float* __restrict__ a_self, const float* __restrict__ a_neigh,
    _Float16* __restrict__ linT, float* __restrict__ s_self,
    float* __restrict__ s_neigh)
{
    const int bh    = blockIdx.x >> 3;       // 0..15
    const int split = blockIdx.x & 7;
    const int b     = bh / GH;
    const int h     = bh % GH;
    const int tid   = threadIdx.x;
    const int wv    = tid >> 5;
    const int lane  = tid & 31;
    const int lh    = lane & 15;
    const bool hiw  = lane >= 16;

    // --- preload W as WMMA B-operands: wb[kt][nt], element e -> K = kt*32 + (hiw?16:0) + e,
    //     col = nt*16 + lh. W is [H,F,F_] row-major fp32.
    const float* Wh = W + (size_t)h * GF * GFO;
    v16h wb[2][4];
    #pragma unroll
    for (int kt = 0; kt < 2; ++kt) {
        const int kbase = kt * 32 + (hiw ? 16 : 0);
        #pragma unroll
        for (int nt = 0; nt < 4; ++nt) {
            const int col = nt * 16 + lh;
            #pragma unroll
            for (int e = 0; e < 16; ++e)
                wb[kt][nt][e] = (_Float16)Wh[(size_t)(kbase + e) * GFO + col];
        }
    }
    float asv[4], anv[4];
    #pragma unroll
    for (int nt = 0; nt < 4; ++nt) {
        asv[nt] = a_self[h * GFO + nt * 16 + lh];
        anv[nt] = a_neigh[h * GFO + nt * 16 + lh];
    }

    const int g = split * 4 + wv;            // 0..31 wave-slot within (b,h)
    for (int mt = g; mt < NT16; mt += 32) {
        const int row0 = mt * 16;

        // --- A-operand: X rows, f32 -> f16. Lane's row = row0 + lh; K base = kt*32 + (hiw?16:0).
        const float* Xr = X + ((size_t)b * GN + row0 + lh) * GF;
        v16h xa[2];
        #pragma unroll
        for (int kt = 0; kt < 2; ++kt) {
            const int kb = kt * 32 + (hiw ? 16 : 0);
            v4f g0 = *(const v4f*)(Xr + kb);
            v4f g1 = *(const v4f*)(Xr + kb + 4);
            v4f g2 = *(const v4f*)(Xr + kb + 8);
            v4f g3 = *(const v4f*)(Xr + kb + 12);
            #pragma unroll
            for (int i = 0; i < 4; ++i) {
                xa[kt][i]      = (_Float16)g0[i];
                xa[kt][4 + i]  = (_Float16)g1[i];
                xa[kt][8 + i]  = (_Float16)g2[i];
                xa[kt][12 + i] = (_Float16)g3[i];
            }
        }

        v8f acc[4] = {};
        #pragma unroll
        for (int kt = 0; kt < 2; ++kt)
            #pragma unroll
            for (int nt = 0; nt < 4; ++nt)
                acc[nt] = __builtin_amdgcn_wmma_f32_16x16x32_f16(
                    false, xa[kt], false, wb[kt][nt], (short)0, acc[nt], false, false);

        // --- store linT (f16, [bh, col, n]) and accumulate s_self/s_neigh partials.
        // C layout: acc[nt][r] -> row = row0 + r + (hiw?8:0), col = nt*16 + lh.
        float ssp[8], snp[8];
        #pragma unroll
        for (int r = 0; r < 8; ++r) { ssp[r] = 0.0f; snp[r] = 0.0f; }

        #pragma unroll
        for (int nt = 0; nt < 4; ++nt) {
            const int col = nt * 16 + lh;
            _Float16* lt = linT + ((size_t)bh * GFO + col) * GN;
            #pragma unroll
            for (int r = 0; r < 8; ++r) {
                const float v = acc[nt][r];
                const int row = row0 + r + (hiw ? 8 : 0);
                lt[row] = (_Float16)v;
                ssp[r] += v * asv[nt];
                snp[r] += v * anv[nt];
            }
        }
        // reduce across the 16 lanes of each half-wave (xor masks 1,2,4,8 keep bit4 fixed)
        #pragma unroll
        for (int r = 0; r < 8; ++r) {
            float s = ssp[r], n = snp[r];
            #pragma unroll
            for (int off = 1; off < 16; off <<= 1) {
                s += __shfl_xor(s, off, 32);
                n += __shfl_xor(n, off, 32);
            }
            if (lh == r) {   // lane r stores row0+r, lane r+16 stores row0+r+8
                const int row = row0 + r + (hiw ? 8 : 0);
                s_self[(size_t)bh * GN + row]  = s;
                s_neigh[(size_t)bh * GN + row] = n;
            }
        }
    }
}

// ---------------------------------------------------------------------------
// Kernel 2: Mneigh[bh] = max_j s_neigh[bh, j]   (softmax shift upper bound)
// ---------------------------------------------------------------------------
__global__ __launch_bounds__(256) void gat_max_kernel(
    const float* __restrict__ s_neigh, float* __restrict__ Mneigh)
{
    const int bh = blockIdx.x;
    float m = -3.0e38f;
    for (int j = threadIdx.x; j < GN; j += 256)
        m = fmaxf(m, s_neigh[(size_t)bh * GN + j]);
    #pragma unroll
    for (int off = 16; off >= 1; off >>= 1)
        m = fmaxf(m, __shfl_xor(m, off, 32));
    __shared__ float sm[8];
    const int wv = threadIdx.x >> 5, ln = threadIdx.x & 31;
    if (ln == 0) sm[wv] = m;
    __syncthreads();
    if (threadIdx.x == 0) {
        float mm = sm[0];
        #pragma unroll
        for (int i = 1; i < 8; ++i) mm = fmaxf(mm, sm[i]);
        Mneigh[bh] = mm;
    }
}

// ---------------------------------------------------------------------------
// Kernel 3: fused masked-softmax attention + aggregation.
// One block per (b, 16-row tile); wave w handles head w.
// Streams A in 16x32 tiles; P built in registers (f16, p<=1 by construction),
// acc += P @ linT-tile via 4x v_wmma_f32_16x16x32_f16 per step.
// ---------------------------------------------------------------------------
__global__ __launch_bounds__(128) void gat_attn_kernel(
    const float* __restrict__ A, const float* __restrict__ s_self,
    const float* __restrict__ s_neigh, const float* __restrict__ Mneigh,
    const _Float16* __restrict__ linT, float* __restrict__ out)
{
    const int blk  = blockIdx.x;
    const int b    = blk / NT16;
    const int mt   = blk % NT16;
    const int row0 = mt * 16;
    const int h    = threadIdx.x >> 5;    // wave = head
    const int lane = threadIdx.x & 31;
    const int lh   = lane & 15;
    const bool hiw = lane >= 16;
    const int bh   = b * GH + h;

    const float s0 = s_self[(size_t)bh * GN + row0 + lh];        // this lane's row
    const float c  = leaky02(s0 + Mneigh[bh]);                   // exact upper bound of row logits

    const float* Arow = A + ((size_t)b * GN + row0 + lh) * GN;   // lane's adjacency row
    const float* sn   = s_neigh + (size_t)bh * GN;
    const int kbA = hiw ? 16 : 0;   // A-operand K range:  kbA..kbA+15 (contiguous)
    const int kbB = hiw ? 16 : 0;   // B-operand K range:  kbB..kbB+15 (contiguous)

    v8f acc[4] = {};
    float lsum = 0.0f;

    for (int j0 = 0; j0 < GN; j0 += 32) {
        __builtin_prefetch(Arow + j0 + 64, 0, 0);   // global_prefetch_b8 next-next tile

        v4f a0 = *(const v4f*)(Arow + j0 + kbA);
        v4f a1 = *(const v4f*)(Arow + j0 + kbA + 4);
        v4f a2 = *(const v4f*)(Arow + j0 + kbA + 8);
        v4f a3 = *(const v4f*)(Arow + j0 + kbA + 12);
        v4f n0 = *(const v4f*)(sn + j0 + kbA);
        v4f n1 = *(const v4f*)(sn + j0 + kbA + 4);
        v4f n2 = *(const v4f*)(sn + j0 + kbA + 8);
        v4f n3 = *(const v4f*)(sn + j0 + kbA + 12);

        float av[16], nv[16];
        #pragma unroll
        for (int i = 0; i < 4; ++i) {
            av[i] = a0[i];      nv[i] = n0[i];
            av[4 + i] = a1[i];  nv[4 + i] = n1[i];
            av[8 + i] = a2[i];  nv[8 + i] = n2[i];
            av[12 + i] = a3[i]; nv[12 + i] = n3[i];
        }

        v16h pa;
        #pragma unroll
        for (int e = 0; e < 16; ++e) {
            const float logit = leaky02(s0 + nv[e]);
            const float p = (av[e] != 0.0f) ? __expf(logit - c) : 0.0f;  // p in [0,1]
            lsum += p;
            pa[e] = (_Float16)p;
        }

        #pragma unroll
        for (int nt = 0; nt < 4; ++nt) {
            const _Float16* lt =
                linT + ((size_t)bh * GFO + nt * 16 + lh) * GN + j0 + kbB;
            const v16h bb = *(const v16h*)lt;   // 32B contiguous: K = kbB..kbB+15
            acc[nt] = __builtin_amdgcn_wmma_f32_16x16x32_f16(
                false, pa, false, bb, (short)0, acc[nt], false, false);
        }
    }

    // row sum: lane m holds K-subset partial for row m; lane m+16 the complement.
    lsum += __shfl_xor(lsum, 16, 32);
    const float linv = 1.0f / lsum;             // valid in all lanes; row = row0 + lh

    // out[b, row, h*F_ + col] = relu(acc) / l
    #pragma unroll
    for (int r = 0; r < 8; ++r) {
        const float sc = __shfl(linv, hiw ? (r + 8) : r, 32);
        const int row = row0 + r + (hiw ? 8 : 0);
        float* orow = out + ((size_t)b * GN + row) * (GH * GFO) + h * GFO;
        #pragma unroll
        for (int nt = 0; nt < 4; ++nt) {
            float v = acc[nt][r];
            v = v > 0.0f ? v * sc : 0.0f;
            orow[nt * 16 + lh] = v;
        }
    }
}

// ---------------------------------------------------------------------------
extern "C" void kernel_launch(void* const* d_in, const int* in_sizes, int n_in,
                              void* d_out, int out_size, void* d_ws, size_t ws_size,
                              hipStream_t stream) {
    (void)in_sizes; (void)n_in; (void)out_size; (void)ws_size;
    const float* X       = (const float*)d_in[0];   // [B,N,F]
    const float* A       = (const float*)d_in[1];   // [B,N,N]
    const float* W       = (const float*)d_in[2];   // [H,F,F_]
    const float* a_self  = (const float*)d_in[3];   // [H,F_]
    const float* a_neigh = (const float*)d_in[4];   // [H,F_]
    float* out = (float*)d_out;                     // [B,N,H*F_]

    char* ws = (char*)d_ws;
    _Float16* linT   = (_Float16*)ws;                                   // 4 MB
    float*    s_self = (float*)(ws + (size_t)4 * 1024 * 1024);          // 128 KB
    float*    s_ngh  = (float*)(ws + (size_t)4 * 1024 * 1024 + 131072); // 128 KB
    float*    Mn     = (float*)(ws + (size_t)4 * 1024 * 1024 + 262144); // 64 B

    gat_lin_kernel<<<dim3(GB * GH * 8), dim3(128), 0, stream>>>(
        X, W, a_self, a_neigh, linT, s_self, s_ngh);
    gat_max_kernel<<<dim3(GB * GH), dim3(256), 0, stream>>>(s_ngh, Mn);
    gat_attn_kernel<<<dim3(GB * NT16), dim3(128), 0, stream>>>(
        A, s_self, s_ngh, Mn, linT, out);
}
